// GCN_90993177133181
// MI455X (gfx1250) — matrix operand order repaired and checked by reading, not compiled, not run on hardware
//
#include <hip/hip_runtime.h>

typedef __attribute__((ext_vector_type(16))) _Float16 v16h;
typedef __attribute__((ext_vector_type(8)))  _Float16 v8h;
typedef __attribute__((ext_vector_type(8)))  float    v8f;

// D = A(16x32 f16) * B(32x16 f16) + C(16x16 f32)
static __device__ __forceinline__ v8f wmma_f16(v16h a, v16h b, v8f c) {
    return __builtin_amdgcn_wmma_f32_16x16x32_f16(false, a, false, b, (short)0, c,
                                                  false, false);
}

// ---------------------------------------------------------------------------
// Weight pre-swizzle into exact WMMA B-operand lane layout (f16).
// B k order chosen here, same as A (16-bit, KxN=32x16): lane L, element e -> B[K = 16*(L/16)+e][N = L%16]
// w1b index: (((ntile*8 + chunk)*32 + lane)*16 + e), k = 32*chunk + 16*(L/16) + e
// ---------------------------------------------------------------------------
__global__ void prep_w1_kernel(const float* __restrict__ W1, _Float16* __restrict__ w1b) {
    int tid = blockIdx.x * 256 + threadIdx.x;            // 0..32767
    int e     = tid & 15;
    int lane  = (tid >> 4) & 31;
    int chunk = (tid >> 9) & 7;
    int ntile = tid >> 12;                               // 0..7
    int k = 32 * chunk + 16 * (lane >> 4) + e;           // 0..255
    int n = 16 * ntile + (lane & 15);                    // 0..127
    w1b[tid] = (_Float16)W1[k * 128 + n];
}

// W2 is 128x40, padded to 48 output cols (3 N-tiles), zero-filled padding.
__global__ void prep_w2_kernel(const float* __restrict__ W2, _Float16* __restrict__ w2b) {
    int tid = blockIdx.x * 256 + threadIdx.x;            // 0..6143
    int e     = tid & 15;
    int lane  = (tid >> 4) & 31;
    int chunk = (tid >> 9) & 3;
    int ntile = tid >> 11;                               // 0..2
    int k = 32 * chunk + 16 * (lane >> 4) + e;           // 0..127
    int n = 16 * ntile + (lane & 15);                    // 0..47
    w2b[tid] = (n < 40) ? (_Float16)W2[k * 40 + n] : (_Float16)0.0f;
}

// ---------------------------------------------------------------------------
// GEMM1: h[100000,128] = x[100000,256] @ W1  (WMMA f16 in, f32 acc)
// block = 256 threads = 8 waves; each block does 16 rows, each wave one N-tile.
// A tile staged in LDS (f32 -> f16), +8 pad -> 528B row stride, conflict-free.
// x is read exactly once -> non-temporal, so L2 stays reserved for h/agg1.
// ---------------------------------------------------------------------------
__global__ __launch_bounds__(256) void gemm1_kernel(const float* __restrict__ x,
                                                    const _Float16* __restrict__ w1b,
                                                    float* __restrict__ h) {
    __shared__ __align__(16) _Float16 ldsA[16][264];     // 16 x (256 + 8 pad)
    const int tid  = threadIdx.x;
    const int row0 = blockIdx.x * 16;

    #pragma unroll
    for (int i = 0; i < 16; ++i)                          // coalesced: col = tid
        ldsA[i][tid] =
            (_Float16)__builtin_nontemporal_load(&x[(size_t)(row0 + i) * 256 + tid]);
    __syncthreads();

    const int wave = tid >> 5;                            // N-tile 0..7
    const int lane = tid & 31;
    const int half = lane >> 4;                           // lane-half for K split
    const int nl   = lane & 15;

    v8f c = {};
    #pragma unroll
    for (int chunk = 0; chunk < 8; ++chunk) {
        const int k0 = chunk * 32;
        // A k order used here (same as w1b): lanes 0-15 hold K {k0..k0+15}, hi-lanes +16
        v8h lo = *(const v8h*)&ldsA[nl][k0 + half * 16];
        v8h hi = *(const v8h*)&ldsA[nl][k0 + half * 16 + 8];
        v16h a;
        #pragma unroll
        for (int e = 0; e < 8; ++e) { a[e] = lo[e]; a[e + 8] = hi[e]; }
        v16h b = *(const v16h*)(w1b + ((((wave << 3) + chunk) << 5) + lane) * 16);
        c = wmma_f16(a, b, c);
    }
    // C/D f32 layout: VGPR j -> M = j + 8*half, N = nl (+wave tile offset)
    #pragma unroll
    for (int j = 0; j < 8; ++j)
        h[(size_t)(row0 + half * 8 + j) * 128 + wave * 16 + nl] = c[j];
}

// ---------------------------------------------------------------------------
// GEMM2: h2[100000,40] = relu(agg1) @ W2, N padded 40->48, 3 waves/block.
// ---------------------------------------------------------------------------
__global__ __launch_bounds__(96) void gemm2_kernel(const float* __restrict__ agg1,
                                                   const _Float16* __restrict__ w2b,
                                                   float* __restrict__ h2) {
    __shared__ __align__(16) _Float16 ldsA[16][136];     // 16 x (128 + 8 pad)
    const int tid  = threadIdx.x;
    const int row0 = blockIdx.x * 16;

    for (int idx = tid; idx < 16 * 128; idx += 96) {
        int r = idx >> 7, cc = idx & 127;
        float v = agg1[(size_t)(row0 + r) * 128 + cc];
        ldsA[r][cc] = (_Float16)fmaxf(v, 0.0f);          // fused ReLU
    }
    __syncthreads();

    const int wave = tid >> 5;                            // N-tile 0..2
    const int lane = tid & 31;
    const int half = lane >> 4;
    const int nl   = lane & 15;

    v8f c = {};
    #pragma unroll
    for (int chunk = 0; chunk < 4; ++chunk) {
        const int k0 = chunk * 32;
        v8h lo = *(const v8h*)&ldsA[nl][k0 + half * 16];
        v8h hi = *(const v8h*)&ldsA[nl][k0 + half * 16 + 8];
        v16h a;
        #pragma unroll
        for (int e = 0; e < 8; ++e) { a[e] = lo[e]; a[e + 8] = hi[e]; }
        v16h b = *(const v16h*)(w2b + ((((wave << 2) + chunk) << 5) + lane) * 16);
        c = wmma_f16(a, b, c);
    }
    const int n = wave * 16 + nl;
    if (n < 40) {
        #pragma unroll
        for (int j = 0; j < 8; ++j)
            h2[(size_t)(row0 + half * 8 + j) * 40 + n] = c[j];
    }
}

// ---------------------------------------------------------------------------
// Edge scatter 1: warp per edge, lane i moves float4 i of 32 (128 floats).
// Source (h, 51MB) and destination (agg1, 51MB) both fit in the 192MB L2,
// so gathers hit L2 and the f32 atomics resolve at L2 rate. The edge index
// stream has zero reuse -> non-temporal loads to avoid evicting h/agg1.
// ---------------------------------------------------------------------------
__global__ __launch_bounds__(256) void scatter1_kernel(const int* __restrict__ src,
                                                       const int* __restrict__ dst,
                                                       const float* __restrict__ h,
                                                       float* __restrict__ agg,
                                                       int E) {
    unsigned gid  = blockIdx.x * 256u + threadIdx.x;
    unsigned edge = gid >> 5;
    if (edge >= (unsigned)E) return;
    int sl = gid & 31;
    int s = __builtin_nontemporal_load(&src[edge]);
    int d = __builtin_nontemporal_load(&dst[edge]);
    float4 v = ((const float4*)(h + (size_t)s * 128))[sl];
    float* o = agg + (size_t)d * 128 + sl * 4;
    atomicAdd(o + 0, v.x);
    atomicAdd(o + 1, v.y);
    atomicAdd(o + 2, v.z);
    atomicAdd(o + 3, v.w);
}

// Edge scatter 2: 10 float4 slots per edge (40 floats).
__global__ __launch_bounds__(256) void scatter2_kernel(const int* __restrict__ src,
                                                       const int* __restrict__ dst,
                                                       const float* __restrict__ h2,
                                                       float* __restrict__ out,
                                                       int E) {
    unsigned gid  = blockIdx.x * 256u + threadIdx.x;
    unsigned edge = gid / 10u;
    if (edge >= (unsigned)E) return;
    int slot = (int)(gid - edge * 10u);
    int s = __builtin_nontemporal_load(&src[edge]);
    int d = __builtin_nontemporal_load(&dst[edge]);
    float4 v = *(const float4*)(h2 + (size_t)s * 40 + slot * 4);
    float* o = out + (size_t)d * 40 + slot * 4;
    atomicAdd(o + 0, v.x);
    atomicAdd(o + 1, v.y);
    atomicAdd(o + 2, v.z);
    atomicAdd(o + 3, v.w);
}

// ---------------------------------------------------------------------------
extern "C" void kernel_launch(void* const* d_in, const int* in_sizes, int n_in,
                              void* d_out, int out_size, void* d_ws, size_t ws_size,
                              hipStream_t stream) {
    const float* x  = (const float*)d_in[0];
    const int*   ei = (const int*)d_in[1];   // [2, E] int32 (JAX default x64-off)
    const float* W1 = (const float*)d_in[2];
    const float* W2 = (const float*)d_in[3];

    const int N = in_sizes[0] / 256;         // 100000
    const int E = in_sizes[1] / 2;           // 1600000
    const int* src = ei;
    const int* dst = ei + E;

    // Workspace layout
    char* ws = (char*)d_ws;
    float*    h    = (float*)ws;                                   // N*128 f32
    float*    agg1 = (float*)(ws + (size_t)N * 128 * 4);           // N*128 f32
    float*    h2   = (float*)(ws + (size_t)N * 128 * 8);           // N*40  f32
    _Float16* w1b  = (_Float16*)(ws + (size_t)N * 128 * 8 + (size_t)N * 40 * 4);
    _Float16* w2b  = (_Float16*)((char*)w1b + 256 * 128 * 2);

    float* outf = (float*)d_out;

    // Zero the scatter accumulators (capture-safe async memsets).
    hipMemsetAsync(agg1, 0, (size_t)N * 128 * 4, stream);
    hipMemsetAsync(outf, 0, (size_t)out_size * 4, stream);

    // Swizzle weights into WMMA B layout (f16).
    prep_w1_kernel<<<(256 * 128) / 256, 256, 0, stream>>>(W1, w1b);
    prep_w2_kernel<<<(3 * 4 * 32 * 16) / 256, 256, 0, stream>>>(W2, w2b);

    // Layer 1
    gemm1_kernel<<<N / 16, 256, 0, stream>>>(x, w1b, h);
    {
        long long tot = (long long)E * 32;
        int blocks = (int)((tot + 255) / 256);
        scatter1_kernel<<<blocks, 256, 0, stream>>>(src, dst, h, agg1, E);
    }

    // Layer 2 (ReLU fused into GEMM2's LDS staging)
    gemm2_kernel<<<N / 16, 96, 0, stream>>>(agg1, w2b, h2);
    {
        long long tot = (long long)E * 10;
        int blocks = (int)((tot + 255) / 256);
        scatter2_kernel<<<blocks, 256, 0, stream>>>(src, dst, h2, outf, E);
    }
}
